// DynamicGRU_85023172591831
// MI455X (gfx1250) — matrix operand order repaired and checked by reading, not compiled, not run on hardware
//
#include <hip/hip_runtime.h>
#include <hip/hip_bf16.h>

typedef __attribute__((ext_vector_type(16))) _Float16 v16h;
typedef __attribute__((ext_vector_type(8)))  _Float16 v8h;
typedef __attribute__((ext_vector_type(8)))  float    v8f;
typedef __attribute__((ext_vector_type(4)))  float    v4f;

#define DEMB 64          // embedding size D
#define KIN  128         // concat input size 2*D
#define ROWP 136         // padded LDS row length in halves (136*2=272B, 16B aligned, breaks bank conflicts)

// Fast activations: v_exp_f32 + v_rcp_f32 (TRANS pipe, co-executes with WMMA),
// avoiding the IEEE div expansion (v_div_scale/v_div_fmas chains).
__device__ __forceinline__ float sigmoid_f(float x) {
    float e = __expf(-x);
    return __builtin_amdgcn_rcpf(1.0f + e);
}
__device__ __forceinline__ float tanh_f(float x) {
    float e = __expf(-2.0f * x);
    return (1.0f - e) * __builtin_amdgcn_rcpf(1.0f + e);
}

union V16U { v16h v; v8h h[2]; };

// A-fragment (16x32 f16, MxK) per ISA 7.12.2:
//   lane m = lane&15, half = lane>>4
//   VGPR 0-3: K = 8*half + 0..7 ; VGPR 4-7: K = 16 + 8*half + 0..7
__device__ __forceinline__ v16h load_a_frag(const _Float16* As, int m, int kc, int half) {
    const _Float16* p = As + m * ROWP + kc * 32 + 8 * half;
    V16U u;
    u.h[0] = *(const v8h*)(p);
    u.h[1] = *(const v8h*)(p + 16);
    return u.v;
}

// B-fragment (32x16 f16, KxN) per ISA 7.12.2 (striped rows):
//   lane n = lane&15, half = lane>>4; VGPR 0-3: K = 8*half + 0..7 ; VGPR 4-7: K = 16 + 8*half + 0..7
//   WT is stored transposed [N][K] so this is 2 contiguous b128 reads.
__device__ __forceinline__ v16h load_b_frag(const _Float16* WT, int n, int kc, int half) {
    const _Float16* p = WT + n * ROWP + kc * 32 + 8 * half;
    V16U u;
    u.h[0] = *(const v8h*)(p);
    u.h[1] = *(const v8h*)(p + 16);
    return u.v;
}

__global__ void __launch_bounds__(32)
augru_scan_kernel(const float* __restrict__ X,      // [B, T, D]
                  const int*   __restrict__ SEQ,    // [B]
                  const float* __restrict__ ATT,    // [B, T]
                  const float* __restrict__ Wg,     // [2D, 2D] row-major (k, n)
                  const float* __restrict__ bg,     // [2D]
                  const float* __restrict__ Wc,     // [2D, D]
                  const float* __restrict__ bc,     // [D]
                  float*       __restrict__ OUT,    // [B, T, D]
                  int T) {
    __shared__ alignas(16) _Float16 WgT[KIN * ROWP];   // transposed gate weights  [n=0..127][k=0..127]
    __shared__ alignas(16) _Float16 WcT[DEMB * ROWP];  // transposed cand weights  [n=0..63][k=0..127]
    __shared__ alignas(16) _Float16 As[16 * ROWP];     // A tile: cols 0..63 = x_t, cols 64..127 = h (or r*h)
    __shared__ float bg_s[KIN];
    __shared__ float bc_s[DEMB];
    __shared__ float att_s[16];
    __shared__ int   seq_s[16];

    const int tid  = threadIdx.x;       // 0..31, one wave per block
    const int half = tid >> 4;          // 0 | 1
    const int ln   = tid & 15;
    const int b0   = blockIdx.x * 16;   // batch rows of this tile

    // ---- stage weights (transposed, f16) + biases into LDS ----
    for (int i = tid; i < KIN * KIN; i += 32) {
        int k = i >> 7, n = i & 127;                 // Wg[k][n]
        WgT[n * ROWP + k] = (_Float16)Wg[i];
    }
    for (int i = tid; i < KIN * DEMB; i += 32) {
        int k = i >> 6, n = i & 63;                  // Wc[k][n]
        WcT[n * ROWP + k] = (_Float16)Wc[i];
    }
    for (int i = tid; i < KIN; i += 32)  bg_s[i] = bg[i];
    for (int i = tid; i < DEMB; i += 32) bc_s[i] = bc[i];
    if (half == 0) seq_s[ln] = SEQ[b0 + ln];

    // zero initial h region of A tile (cols 64..127): lane handles row ln, 32 cols
    {
        _Float16* dst = As + ln * ROWP + 64 + 32 * half;
        v8h z = {};
        #pragma unroll
        for (int c = 0; c < 32; c += 8) *(v8h*)(dst + c) = z;
    }
    __syncthreads();

    // h state in C/D fragment layout: hcd[j] covers cols 16j..16j+15;
    // VGPR r of a fragment holds row (r + 8*half), col (16j + ln).
    v8f hcd[4];
    #pragma unroll
    for (int j = 0; j < 4; j++) hcd[j] = (v8f){};

    for (int t = 0; t < T; t++) {
        // ---- load x_t tile into A cols 0..63 (f32 -> f16); lane: row ln, cols 32*half..+31 ----
        {
            const float* xr = X + ((size_t)(b0 + ln) * T + t) * DEMB + 32 * half;
            _Float16* dst = As + ln * ROWP + 32 * half;
            #pragma unroll
            for (int c = 0; c < 32; c += 8) {
                v4f f0 = *(const v4f*)(xr + c);
                v4f f1 = *(const v4f*)(xr + c + 4);
                v8h hh;
                #pragma unroll
                for (int i = 0; i < 4; i++) { hh[i] = (_Float16)f0[i]; hh[4 + i] = (_Float16)f1[i]; }
                *(v8h*)(dst + c) = hh;
            }
        }
        if (half == 0) att_s[ln] = ATT[(size_t)(b0 + ln) * T + t];
        asm volatile("" ::: "memory");

        // ---- gate GEMM: [16,128] @ [128,128] -> [16,128] ----
        v16h afr[4];
        #pragma unroll
        for (int kc = 0; kc < 4; kc++) afr[kc] = load_a_frag(As, ln, kc, half);

        v8f accg[8];
        #pragma unroll
        for (int j = 0; j < 8; j++) {
            float bias = bg_s[16 * j + ln];
            v8f acc;
            #pragma unroll
            for (int r = 0; r < 8; r++) acc[r] = bias;
            #pragma unroll
            for (int kc = 0; kc < 4; kc++) {
                v16h bf = load_b_frag(WgT, 16 * j + ln, kc, half);
                acc = __builtin_amdgcn_wmma_f32_16x16x32_f16(
                          false, afr[kc], false, bf, (short)0, acc, false, false);
            }
            accg[j] = acc;
        }

        // ---- gates: r = sigmoid(cols 0..63), u = sigmoid(cols 64..127) ----
        v8f rg[4], ug[4];
        #pragma unroll
        for (int j = 0; j < 4; j++) {
            #pragma unroll
            for (int r = 0; r < 8; r++) {
                rg[j][r] = sigmoid_f(accg[j][r]);
                ug[j][r] = sigmoid_f(accg[4 + j][r]);
            }
        }

        // ---- write r*h (f16) into A cols 64..127 for candidate GEMM ----
        #pragma unroll
        for (int j = 0; j < 4; j++) {
            #pragma unroll
            for (int r = 0; r < 8; r++) {
                int row = r + 8 * half;
                As[row * ROWP + 64 + 16 * j + ln] = (_Float16)(rg[j][r] * hcd[j][r]);
            }
        }
        asm volatile("" ::: "memory");

        // ---- candidate GEMM: [16,128] @ [128,64] -> [16,64] ----
        #pragma unroll
        for (int kc = 2; kc < 4; kc++) afr[kc] = load_a_frag(As, ln, kc, half);  // x part (kc 0,1) reused

        v8f accc[4];
        #pragma unroll
        for (int j = 0; j < 4; j++) {
            float bias = bc_s[16 * j + ln];
            v8f acc;
            #pragma unroll
            for (int r = 0; r < 8; r++) acc[r] = bias;
            #pragma unroll
            for (int kc = 0; kc < 4; kc++) {
                v16h bf = load_b_frag(WcT, 16 * j + ln, kc, half);
                acc = __builtin_amdgcn_wmma_f32_16x16x32_f16(
                          false, afr[kc], false, bf, (short)0, acc, false, false);
            }
            accc[j] = acc;
        }

        // ---- elementwise AUGRU update + masking + output + next-step h staging ----
        #pragma unroll
        for (int j = 0; j < 4; j++) {
            #pragma unroll
            for (int r = 0; r < 8; r++) {
                int row = r + 8 * half;
                float c  = tanh_f(accc[j][r]);
                float ua = att_s[row] * ug[j][r];
                float hn = (1.0f - ua) * hcd[j][r] + ua * c;
                bool valid = (t < seq_s[row]);
                float hv = valid ? hn : hcd[j][r];
                hcd[j][r] = hv;
                OUT[((size_t)(b0 + row) * T + t) * DEMB + 16 * j + ln] = valid ? hn : 0.0f;
                As[row * ROWP + 64 + 16 * j + ln] = (_Float16)hv;   // h for next step's gate GEMM
            }
        }
        asm volatile("" ::: "memory");
    }
}

extern "C" void kernel_launch(void* const* d_in, const int* in_sizes, int n_in,
                              void* d_out, int out_size, void* d_ws, size_t ws_size,
                              hipStream_t stream) {
    const float* X   = (const float*)d_in[0];   // rnn_input [B,T,D] f32
    const int*   SEQ = (const int*)  d_in[1];   // sequence_length [B,1] i32
    const float* ATT = (const float*)d_in[2];   // att_score [B,T,1] f32
    const float* Wg  = (const float*)d_in[3];   // W_gate [2D,2D] f32
    const float* bg  = (const float*)d_in[4];   // b_gate [2D]
    const float* Wc  = (const float*)d_in[5];   // W_cand [2D,D]
    const float* bc  = (const float*)d_in[6];   // b_cand [D]
    float* OUT = (float*)d_out;

    int B = in_sizes[1];            // sequence_length has B elements
    int T = in_sizes[2] / B;        // att_score has B*T elements

    dim3 grid(B / 16);              // one wave32 per 16-row batch tile
    dim3 block(32);
    augru_scan_kernel<<<grid, block, 0, stream>>>(X, SEQ, ATT, Wg, bg, Wc, bc, OUT, T);
}
